// VisionMamba_7292854468980
// MI455X (gfx1250) — matrix operand order repaired and checked by reading, not compiled, not run on hardware
//
#include <hip/hip_runtime.h>
#include <hip/hip_bf16.h>

// ---------------------------------------------------------------------------
// Vision Mamba forward for MI455X (gfx1250, wave32).
// All GEMMs use v_wmma_f32_16x16x32_f16 (f32 accumulate), fragments built per
// CDNA5 ISA 7.12.2 layouts. Elementwise / scan stages are fused f32 kernels.
// ---------------------------------------------------------------------------

typedef _Float16 f16;
typedef _Float16 v16h __attribute__((ext_vector_type(16)));
typedef _Float16 v8h  __attribute__((ext_vector_type(8)));
typedef float    v8f  __attribute__((ext_vector_type(8)));

#define B_     32
#define L_     196
#define DM     384
#define DEPTH_ 24
#define DI     768
#define NS     16
#define DTR    24
#define MTOK   (B_ * L_)          // 6272 tokens (= 392 * 16)

// ---------------------------------------------------------------------------
// Generic WMMA GEMM:  C[M,Nstore] = act(A[M,K] * W[Nalloc,K]^T + bias)
//   A, W are f16 row-major; C is f32 with leading dim ldc.
//   Each wave computes a 16(M) x 64(N) strip: 4 wmma tiles sharing one A frag.
//   act: 0 = none, 1 = softplus (for Mamba delta).
// ---------------------------------------------------------------------------
__global__ __launch_bounds__(256)
void gemm_wmma_f16(const f16* __restrict__ A, const f16* __restrict__ W,
                   float* __restrict__ C, int M, int Nalloc, int K,
                   int ldc, int Nstore, const float* __restrict__ bias, int act)
{
    const int lane = threadIdx.x & 31;
    const int wave = threadIdx.x >> 5;
    const int tilesN4 = Nalloc >> 6;               // strips of 64 columns
    const int tile = blockIdx.x * 8 + wave;
    if (tile >= (M >> 4) * tilesN4) return;
    const int tm  = tile / tilesN4;
    const int tn4 = tile % tilesN4;

    const int l15  = lane & 15;
    const int hi   = (lane & 16) ? 1 : 0;
    const int kA   = hi ? 8 : 0;                   // A frag K base (ISA 7.12.2)
    const int kB   = hi ? 8 : 0;                   // B frag K base
    const int arow = (tm * 16 + l15) * K;

    v8f acc[4];
    #pragma unroll
    for (int j = 0; j < 4; ++j) acc[j] = (v8f){};

    for (int k0 = 0; k0 < K; k0 += 32) {
        // A fragment: halves 0..7 -> K = k0+kA+i ; halves 8..15 -> K = k0+16+kA+i
        v8h aLo = *(const v8h*)(A + arow + k0 + kA);
        v8h aHi = *(const v8h*)(A + arow + k0 + kA + 16);
        v16h af = __builtin_shufflevector(aLo, aHi,
                    0,1,2,3,4,5,6,7,8,9,10,11,12,13,14,15);
        if (k0 + 32 < K) {
            __builtin_prefetch(A + arow + k0 + 32 + kA, 0, 1);  // global_prefetch_b8
        }
        #pragma unroll
        for (int j = 0; j < 4; ++j) {
            const int n = tn4 * 64 + j * 16 + l15;
            const int wrow = n * K + k0 + kB;       // B frag: same K order as A
            v8h bLo = *(const v8h*)(W + wrow);
            v8h bHi = *(const v8h*)(W + wrow + 16);
            v16h bf = __builtin_shufflevector(bLo, bHi,
                        0,1,2,3,4,5,6,7,8,9,10,11,12,13,14,15);
            acc[j] = __builtin_amdgcn_wmma_f32_16x16x32_f16(
                         false, af, false, bf, (short)0, acc[j], false, false);
        }
    }

    // C/D layout: VGPR r -> row = tm*16 + r + 8*hi, col = lane&15 (ISA 7.12.2)
    const int rbase = tm * 16 + hi * 8;
    #pragma unroll
    for (int j = 0; j < 4; ++j) {
        const int n = tn4 * 64 + j * 16 + l15;
        if (n < Nstore) {
            const float bv = bias ? bias[n] : 0.0f;
            #pragma unroll
            for (int r = 0; r < 8; ++r) {
                float v = acc[j][r] + bv;
                if (act == 1) v = (v > 20.0f) ? v : log1pf(__expf(v));
                C[(rbase + r) * ldc + n] = v;
            }
        }
    }
}

// ---------------------------------------------------------------------------
// Weight conversion f32 -> f16 (flat, and padded variants)
// ---------------------------------------------------------------------------
__global__ void cvt_f16_kernel(const float* __restrict__ src,
                               f16* __restrict__ dst, int n)
{
    int i = blockIdx.x * blockDim.x + threadIdx.x;
    if (i < n) dst[i] = (f16)src[i];
}

__global__ void cvt_pad_kernel(const float* __restrict__ src,
                               f16* __restrict__ dst,
                               int srcRows, int srcCols, int dstCols, int total)
{
    int i = blockIdx.x * blockDim.x + threadIdx.x;
    if (i >= total) return;
    int c = i % dstCols, r = i / dstCols;
    dst[i] = (r < srcRows && c < srcCols) ? (f16)src[r * srcCols + c] : (f16)0.0f;
}

// ---------------------------------------------------------------------------
// Patch embed im2col: token m = b*196 + (ph*14+pw); k = c*256 + py*16 + px.
// Matches patch_w flat layout [384][3][16][16] -> GEMM weight [384, 768].
// ---------------------------------------------------------------------------
__global__ void im2col_kernel(const float* __restrict__ x, f16* __restrict__ out)
{
    int idx = blockIdx.x * blockDim.x + threadIdx.x;
    if (idx >= MTOK * 768) return;
    int k = idx % 768, m = idx / 768;
    int b = m / L_, l = m % L_;
    int ph = l / 14, pw = l % 14;
    int c = k >> 8, r = k & 255;
    int py = r >> 4, px = r & 15;
    out[idx] = (f16)x[((b * 3 + c) * 224 + ph * 16 + py) * 224 + pw * 16 + px];
}

__global__ void add_pos_init_kernel(float* __restrict__ hidden,
                                    float* __restrict__ residual,
                                    const float* __restrict__ pos)
{
    int idx = blockIdx.x * blockDim.x + threadIdx.x;
    if (idx >= MTOK * DM) return;
    int dcol = idx % DM, m = idx / DM, l = m % L_;
    hidden[idx] += pos[l * DM + dcol];
    residual[idx] = 0.0f;
}

// ---------------------------------------------------------------------------
// residual += hidden ; out = f16(LayerNorm(residual) * w + b). One block/token.
// ---------------------------------------------------------------------------
__global__ __launch_bounds__(128)
void add_ln_kernel(const float* __restrict__ hidden, float* __restrict__ residual,
                   const float* __restrict__ w, const float* __restrict__ bb,
                   f16* __restrict__ outH)
{
    __shared__ float red[128];
    const int m = blockIdx.x;
    const int t = threadIdx.x;
    float v[3];
    float s = 0.0f;
    #pragma unroll
    for (int i = 0; i < 3; ++i) {
        int c = t + i * 128;
        float x = residual[m * DM + c] + hidden[m * DM + c];
        residual[m * DM + c] = x;
        v[i] = x; s += x;
    }
    red[t] = s; __syncthreads();
    for (int o = 64; o > 0; o >>= 1) { if (t < o) red[t] += red[t + o]; __syncthreads(); }
    const float mu = red[0] * (1.0f / DM);
    __syncthreads();
    float sq = 0.0f;
    #pragma unroll
    for (int i = 0; i < 3; ++i) { float d = v[i] - mu; sq += d * d; }
    red[t] = sq; __syncthreads();
    for (int o = 64; o > 0; o >>= 1) { if (t < o) red[t] += red[t + o]; __syncthreads(); }
    const float rstd = rsqrtf(red[0] * (1.0f / DM) + 1e-5f);
    #pragma unroll
    for (int i = 0; i < 3; ++i) {
        int c = t + i * 128;
        outH[m * DM + c] = (f16)((v[i] - mu) * rstd * w[c] + bb[c]);
    }
}

// ---------------------------------------------------------------------------
// Depthwise causal conv (k=4) over sequence + bias + SiLU. Reads xc half of xz.
// ---------------------------------------------------------------------------
__global__ void dwconv_silu_kernel(const float* __restrict__ xz,
                                   const float* __restrict__ cw,  // [DI,4]
                                   const float* __restrict__ cb,  // [DI]
                                   float* __restrict__ xc, f16* __restrict__ xcH)
{
    int idx = blockIdx.x * blockDim.x + threadIdx.x;
    if (idx >= MTOK * DI) return;
    int d = idx % DI, m = idx / DI;
    int l = m % L_, base = m - l;
    float acc = cb[d];
    #pragma unroll
    for (int j = 0; j < 4; ++j) {
        int ll = l - 3 + j;
        if (ll >= 0) acc += xz[(base + ll) * (2 * DI) + d] * cw[d * 4 + j];
    }
    float sv = acc / (1.0f + __expf(-acc));
    xc[idx]  = sv;
    xcH[idx] = (f16)sv;
}

// dt slice of proj (cols 0..23) -> f16 padded to 32 (K for delta GEMM)
__global__ void dt_extract_kernel(const float* __restrict__ proj,
                                  f16* __restrict__ dtH)
{
    int idx = blockIdx.x * blockDim.x + threadIdx.x;
    if (idx >= MTOK * 32) return;
    int c = idx & 31, r = idx >> 5;
    dtH[idx] = (c < DTR) ? (f16)proj[r * 64 + c] : (f16)0.0f;
}

// ---------------------------------------------------------------------------
// Selective scan: one thread per (b, d). B_t/C_t rows broadcast through LDS.
// Fuses D-residual and SiLU(z) gate; writes y directly as f16 for out-proj.
// ---------------------------------------------------------------------------
__global__ __launch_bounds__(256)
void scan_kernel(const float* __restrict__ proj,   // [MTOK,64] (dt|B|C)
                 const float* __restrict__ delta,  // [MTOK,DI]
                 const float* __restrict__ xc,     // [MTOK,DI]
                 const float* __restrict__ xz,     // [MTOK,2*DI] (z at +DI)
                 const float* __restrict__ A_log,  // [DI,NS] layer slice
                 const float* __restrict__ Dp,     // [DI]
                 f16* __restrict__ yH)             // [MTOK,DI]
{
    __shared__ float sBC[32];
    const int d = blockIdx.x * blockDim.x + threadIdx.x;   // 0..DI-1
    const int b = blockIdx.y;
    float a[NS], h[NS];
    #pragma unroll
    for (int n = 0; n < NS; ++n) { a[n] = -__expf(A_log[d * NS + n]); h[n] = 0.0f; }
    const float Dv = Dp[d];
    for (int l = 0; l < L_; ++l) {
        const int m = b * L_ + l;
        if (threadIdx.x < 32) sBC[threadIdx.x] = proj[m * 64 + DTR + threadIdx.x];
        __syncthreads();
        const float dlt = delta[m * DI + d];
        const float u   = xc[m * DI + d];
        const float zv  = xz[m * 2 * DI + DI + d];
        const float du  = dlt * u;
        float ys = 0.0f;
        #pragma unroll
        for (int n = 0; n < NS; ++n) {
            h[n] = __expf(dlt * a[n]) * h[n] + du * sBC[n];
            ys  += h[n] * sBC[NS + n];
        }
        const float y = (ys + u * Dv) * (zv / (1.0f + __expf(-zv)));
        yH[m * DI + d] = (f16)y;
        __syncthreads();
    }
}

// mean pool over L into f16 for the head GEMM
__global__ void pool_kernel(const f16* __restrict__ lnH, f16* __restrict__ pooled)
{
    int idx = blockIdx.x * blockDim.x + threadIdx.x;
    if (idx >= B_ * DM) return;
    int dcol = idx % DM, b = idx / DM;
    float s = 0.0f;
    for (int l = 0; l < L_; ++l) s += (float)lnH[(b * L_ + l) * DM + dcol];
    pooled[idx] = (f16)(s * (1.0f / L_));
}

// ---------------------------------------------------------------------------
// Host-side orchestration
// ---------------------------------------------------------------------------
static inline int ceil_div(int a, int b) { return (a + b - 1) / b; }

extern "C" void kernel_launch(void* const* d_in, const int* in_sizes, int n_in,
                              void* d_out, int out_size, void* d_ws, size_t ws_size,
                              hipStream_t stream)
{
    (void)in_sizes; (void)n_in; (void)out_size; (void)ws_size;

    const float* x        = (const float*)d_in[0];
    const float* patch_w  = (const float*)d_in[1];
    const float* patch_b  = (const float*)d_in[2];
    const float* pos      = (const float*)d_in[3];
    const float* W_in     = (const float*)d_in[4];
    const float* conv_w   = (const float*)d_in[5];
    const float* conv_b   = (const float*)d_in[6];
    const float* W_xp     = (const float*)d_in[7];
    const float* W_dt     = (const float*)d_in[8];
    const float* b_dt     = (const float*)d_in[9];
    const float* A_log    = (const float*)d_in[10];
    const float* D_p      = (const float*)d_in[11];
    const float* W_out    = (const float*)d_in[12];
    const float* ln_w     = (const float*)d_in[13];
    const float* ln_b     = (const float*)d_in[14];
    const float* normf_w  = (const float*)d_in[15];
    const float* normf_b  = (const float*)d_in[16];
    const float* head_w   = (const float*)d_in[17];
    const float* head_b   = (const float*)d_in[18];
    float* out            = (float*)d_out;

    // ---- workspace layout (256B aligned slices) ----
    char* ws = (char*)d_ws;
    size_t off = 0;
    auto alloc = [&](size_t bytes) -> void* {
        void* p = ws + off;
        off = (off + bytes + 255) & ~(size_t)255;
        return p;
    };
    f16*   wInH    = (f16*)  alloc((size_t)DEPTH_ * 2 * DI * DM * 2);   // [24][1536][384]
    f16*   wOutH   = (f16*)  alloc((size_t)DEPTH_ * DM * DI * 2);       // [24][384][768]
    f16*   wXpH    = (f16*)  alloc((size_t)DEPTH_ * 64 * DI * 2);       // [24][64][768] (56->64)
    f16*   wDtH    = (f16*)  alloc((size_t)DEPTH_ * DI * 32 * 2);       // [24][768][32] (24->32)
    f16*   wPatchH = (f16*)  alloc((size_t)DM * 768 * 2);               // [384][768]
    f16*   wHeadH  = (f16*)  alloc((size_t)1024 * DM * 2);              // [1024][384] (1000->1024)
    f16*   im2cH   = (f16*)  alloc((size_t)MTOK * 768 * 2);
    float* hidden  = (float*)alloc((size_t)MTOK * DM * 4);
    float* residual= (float*)alloc((size_t)MTOK * DM * 4);
    f16*   xlnH    = (f16*)  alloc((size_t)MTOK * DM * 2);
    float* xz      = (float*)alloc((size_t)MTOK * 2 * DI * 4);
    float* xc      = (float*)alloc((size_t)MTOK * DI * 4);
    f16*   xcH     = (f16*)  alloc((size_t)MTOK * DI * 2);
    float* proj    = (float*)alloc((size_t)MTOK * 64 * 4);
    f16*   dtH     = (f16*)  alloc((size_t)MTOK * 32 * 2);
    float* delta   = (float*)alloc((size_t)MTOK * DI * 4);
    f16*   yH      = (f16*)  alloc((size_t)MTOK * DI * 2);
    f16*   pooledH = (f16*)  alloc((size_t)B_ * DM * 2);

    // ---- weight conversions (f32 -> f16, with padding where needed) ----
    {
        int n = DEPTH_ * 2 * DI * DM;
        cvt_f16_kernel<<<ceil_div(n, 256), 256, 0, stream>>>(W_in, wInH, n);
        n = DEPTH_ * DM * DI;
        cvt_f16_kernel<<<ceil_div(n, 256), 256, 0, stream>>>(W_out, wOutH, n);
        n = DM * 768;
        cvt_f16_kernel<<<ceil_div(n, 256), 256, 0, stream>>>(patch_w, wPatchH, n);
        n = 1024 * DM;
        cvt_pad_kernel<<<ceil_div(n, 256), 256, 0, stream>>>(head_w, wHeadH, 1000, DM, DM, n);
        for (int ly = 0; ly < DEPTH_; ++ly) {
            int t1 = 64 * DI;
            cvt_pad_kernel<<<ceil_div(t1, 256), 256, 0, stream>>>(
                W_xp + (size_t)ly * 56 * DI, wXpH + (size_t)ly * 64 * DI, 56, DI, DI, t1);
            int t2 = DI * 32;
            cvt_pad_kernel<<<ceil_div(t2, 256), 256, 0, stream>>>(
                W_dt + (size_t)ly * DI * DTR, wDtH + (size_t)ly * DI * 32, DI, DTR, 32, t2);
        }
    }

    auto gemm = [&](const f16* A, const f16* W, float* C, int M, int Nalloc,
                    int K, int ldc, int Nstore, const float* bias, int act) {
        int tiles = (M / 16) * (Nalloc / 64);
        gemm_wmma_f16<<<ceil_div(tiles, 8), 256, 0, stream>>>(
            A, W, C, M, Nalloc, K, ldc, Nstore, bias, act);
    };

    // ---- patch embedding ----
    {
        int n = MTOK * 768;
        im2col_kernel<<<ceil_div(n, 256), 256, 0, stream>>>(x, im2cH);
        gemm(im2cH, wPatchH, hidden, MTOK, DM, 768, DM, DM, patch_b, 0);
        n = MTOK * DM;
        add_pos_init_kernel<<<ceil_div(n, 256), 256, 0, stream>>>(hidden, residual, pos);
    }

    // ---- 24 Mamba blocks ----
    const int nEl = MTOK * DI;
    for (int ly = 0; ly < DEPTH_; ++ly) {
        add_ln_kernel<<<MTOK, 128, 0, stream>>>(
            hidden, residual, ln_w + ly * DM, ln_b + ly * DM, xlnH);
        gemm(xlnH, wInH + (size_t)ly * 2 * DI * DM, xz,
             MTOK, 2 * DI, DM, 2 * DI, 2 * DI, nullptr, 0);
        dwconv_silu_kernel<<<ceil_div(nEl, 256), 256, 0, stream>>>(
            xz, conv_w + (size_t)ly * DI * 4, conv_b + (size_t)ly * DI, xc, xcH);
        gemm(xcH, wXpH + (size_t)ly * 64 * DI, proj, MTOK, 64, DI, 64, 64, nullptr, 0);
        {
            int n = MTOK * 32;
            dt_extract_kernel<<<ceil_div(n, 256), 256, 0, stream>>>(proj, dtH);
        }
        gemm(dtH, wDtH + (size_t)ly * DI * 32, delta,
             MTOK, DI, 32, DI, DI, b_dt + ly * DI, 1 /*softplus*/);
        scan_kernel<<<dim3(DI / 256, B_), 256, 0, stream>>>(
            proj, delta, xc, xz, A_log + (size_t)ly * DI * NS, D_p + ly * DI, yH);
        gemm(yH, wOutH + (size_t)ly * DM * DI, hidden, MTOK, DM, DI, DM, DM, nullptr, 0);
    }

    // ---- final LN, mean pool, classifier head ----
    add_ln_kernel<<<MTOK, 128, 0, stream>>>(hidden, residual, normf_w, normf_b, xlnH);
    {
        int n = B_ * DM;
        pool_kernel<<<ceil_div(n, 256), 256, 0, stream>>>(xlnH, pooledH);
    }
    gemm(pooledH, wHeadH, out, B_, 1024, DM, 1000, 1000, head_b, 0);
}
